// DTGRULayer_28793460752980
// MI455X (gfx1250) — compile-verified, matrix operand order checked by reading, not run on hardware
//
#include <hip/hip_runtime.h>

// ---------------------------------------------------------------------------
// DTGRU-style recurrence on MI455X (gfx1250, wave32, WMMA bf16 -> f32 acc).
// Per time step: 7 dependent fused WMMA GEMMs + 1 softmax kernel.
// Weights converted+TRANSPOSED to bf16 once per launch (L2-resident, ~7.5MB).
// GEMM block tile 128x128, wave tile 32x64 (2 A-frags x 4 B-frags = 8 WMMA
// per 32-k chunk; each B fragment feeds two WMMAs), register-staged
// double-buffered LDS software pipeline. Gating elementwise ops fused into
// GEMM epilogues.
// ---------------------------------------------------------------------------

typedef __attribute__((ext_vector_type(16))) __bf16 v16bf;
typedef __attribute__((ext_vector_type(8)))  float  v8f;

constexpr int Bsz  = 256;   // batch
constexpr int Tlen = 256;   // sequence length
constexpr int Hd   = 512;   // hidden dim
constexpr int Ein  = 64;    // encoder input dim

constexpr int SA_LD = 40;   // LDS row stride (elems), 32-elem payload + pad
constexpr int SB_LD = 40;

enum { EPI_GATE = 0, EPI_NBLEND = 1, EPI_TANH_B = 2, EPI_NONE = 3,
       EPI_TANH_FB = 4, EPI_HBLEND = 5 };

union FragBF { v16bf v; unsigned int u[8]; __bf16 h[16]; };

__device__ __forceinline__ float sigmoidf_(float x) {
  return 1.0f / (1.0f + __expf(-x));
}

// Fused dual-source WMMA GEMM:
//   OUT = epi( A1[M,K1]@W1 + A2[M,K2]@W2 + bias ),  W given TRANSPOSED [N,K].
// M fixed = 256. Block tile 128x128, 8 waves, each wave a 32x64 strip.
// K1, K2 multiples of 32; dual sources flattened into one pipelined K loop.
template <int EPI>
__global__ __launch_bounds__(256)
void wmma_gemm_fused(const __bf16* __restrict__ A1, int lda1, int K1,
                     const __bf16* __restrict__ WT1,
                     const __bf16* __restrict__ A2, int lda2, int K2,
                     const __bf16* __restrict__ WT2,
                     int N,
                     const float* __restrict__ bias,
                     const float* __restrict__ zread,   // z gate [256x512] (blend)
                     float*  __restrict__ zwrite,       // z gate out (EPI_GATE)
                     const float* __restrict__ rmul,    // N or m fp32 (EPI_GATE)
                     __bf16* __restrict__ rbout,        // N*r / m*r bf16 (EPI_GATE)
                     float*  __restrict__ state,        // h or N fp32 [256x512]
                     __bf16* __restrict__ stateb,       // bf16 mirror of state
                     __bf16* __restrict__ dtb,          // d_t bf16 (EPI_HBLEND)
                     float*  __restrict__ outf,
                     __bf16* __restrict__ outb,
                     float*  __restrict__ seq, int seq_stride)
{
  __shared__ __align__(16) __bf16 sA[2][128 * SA_LD];
  __shared__ __align__(16) __bf16 sB[2][128 * SB_LD];

  const int tid   = threadIdx.x;
  const int lane  = tid & 31;
  const int wid   = tid >> 5;
  const int wm    = wid >> 1;          // 0..3 -> 32-row strip
  const int wn    = wid & 1;           // 0..1 -> 64-col strip
  const int bm    = blockIdx.y * 128;
  const int bn    = blockIdx.x * 128;
  const int khalf = lane >> 4;
  const int l16   = lane & 15;

  // staging coordinates (per thread): both tiles are 128 rows x 32 k
  const int sr = tid >> 2;             // row 0..63 (and +64)
  const int sc = (tid & 3) << 3;       // k offset {0,8,16,24}

  const int KT = K1 + K2;

  v8f acc[2][4];
  #pragma unroll
  for (int ai = 0; ai < 2; ++ai)
    #pragma unroll
    for (int i = 0; i < 4; ++i)
      #pragma unroll
      for (int j = 0; j < 8; ++j) acc[ai][i][j] = 0.0f;

  uint4 ra0, ra1, rb0, rb1;

  // load the (kk)-chunk of A and W tiles into registers; prefetch kk+32
  auto load_tiles = [&](int kk) {
    const __bf16 *ap0, *ap1, *wp0, *wp1;
    bool nxt;
    if (kk < K1) {
      ap0 = A1  + (size_t)(bm + sr) * lda1 + (size_t)(kk + sc);
      ap1 = A1  + (size_t)(bm + sr + 64) * lda1 + (size_t)(kk + sc);
      wp0 = WT1 + (size_t)(bn + sr) * K1 + (size_t)(kk + sc);
      wp1 = WT1 + (size_t)(bn + sr + 64) * K1 + (size_t)(kk + sc);
      nxt = (kk + 32 < K1);
    } else {
      int k2 = kk - K1;
      ap0 = A2  + (size_t)(bm + sr) * lda2 + (size_t)(k2 + sc);
      ap1 = A2  + (size_t)(bm + sr + 64) * lda2 + (size_t)(k2 + sc);
      wp0 = WT2 + (size_t)(bn + sr) * K2 + (size_t)(k2 + sc);
      wp1 = WT2 + (size_t)(bn + sr + 64) * K2 + (size_t)(k2 + sc);
      nxt = (kk + 32 < KT);
    }
    ra0 = *reinterpret_cast<const uint4*>(ap0);
    ra1 = *reinterpret_cast<const uint4*>(ap1);
    rb0 = *reinterpret_cast<const uint4*>(wp0);
    rb1 = *reinterpret_cast<const uint4*>(wp1);
    if (nxt) {                         // prefetch chunk-after-next (same src)
      __builtin_prefetch(wp0 + 32, 0, 3);
      __builtin_prefetch(wp1 + 32, 0, 3);
      __builtin_prefetch(ap0 + 32, 0, 3);
      __builtin_prefetch(ap1 + 32, 0, 3);
    }
  };

  load_tiles(0);
  int cur = 0;

  for (int kk = 0; kk < KT; kk += 32) {
    // ---- commit staged registers to LDS buffer `cur` ----
    *reinterpret_cast<uint4*>(&sA[cur][sr * SA_LD + sc])        = ra0;
    *reinterpret_cast<uint4*>(&sA[cur][(sr + 64) * SA_LD + sc]) = ra1;
    *reinterpret_cast<uint4*>(&sB[cur][sr * SB_LD + sc])        = rb0;
    *reinterpret_cast<uint4*>(&sB[cur][(sr + 64) * SB_LD + sc]) = rb1;
    __syncthreads();

    // ---- issue next chunk's global loads (waited one iter later) ----
    if (kk + 32 < KT) load_tiles(kk + 32);

    // ---- two A fragments (rows wm*32 and wm*32+16) ----
    FragBF a[2];
    #pragma unroll
    for (int ai = 0; ai < 2; ++ai) {
      #pragma unroll
      for (int j = 0; j < 8; ++j) {
        int kp = 2 * (j & 3) + 16 * (j >> 2) + 8 * khalf;
        a[ai].u[j] = *reinterpret_cast<const unsigned int*>(
            &sA[cur][(wm * 32 + ai * 16 + l16) * SA_LD + kp]);
      }
    }
    // ---- 4 B fragments, each feeding two WMMAs ----
    #pragma unroll
    for (int i = 0; i < 4; ++i) {
      FragBF b;
      int cl = wn * 64 + i * 16 + l16;
      #pragma unroll
      for (int j = 0; j < 8; ++j) {
        int kp = 2 * (j & 3) + 16 * (j >> 2) + 8 * khalf;
        b.u[j] = *reinterpret_cast<const unsigned int*>(
            &sB[cur][cl * SB_LD + kp]);
      }
      acc[0][i] = __builtin_amdgcn_wmma_f32_16x16x32_bf16(
          false, a[0].v, false, b.v, (short)0, acc[0][i], false, false);
      acc[1][i] = __builtin_amdgcn_wmma_f32_16x16x32_bf16(
          false, a[1].v, false, b.v, (short)0, acc[1][i], false, false);
    }
    cur ^= 1;
  }

  // ---- fused epilogue (lane (m,n) mapping per ISA C/D layout) ----
  #pragma unroll
  for (int ai = 0; ai < 2; ++ai) {
    #pragma unroll
    for (int i = 0; i < 4; ++i) {
      const int col = bn + wn * 64 + i * 16 + l16;
      #pragma unroll
      for (int r = 0; r < 8; ++r) {
        const int row = bm + wm * 32 + ai * 16 + khalf * 8 + r;
        float v = acc[ai][i][r];
        if constexpr (EPI == EPI_GATE) {
          // gates [256x1024]: cols<512 -> z (fp32); cols>=512 -> bf16(r*rmul)
          float sg = sigmoidf_(v + bias[col]);
          if (col < Hd) {                        // uniform per block (bn%128==0)
            zwrite[(size_t)row * Hd + col] = sg;
          } else {
            size_t o = (size_t)row * Hd + (col - Hd);
            rbout[o] = (__bf16)(sg * rmul[o]);
          }
        } else if constexpr (EPI == EPI_NBLEND) {
          float hat = tanhf(v + bias[col]);
          size_t o  = (size_t)row * Hd + col;
          float z   = zread[o];
          float nn  = (1.0f - z) * state[o] + z * hat;
          state[o]  = nn;
          stateb[o] = (__bf16)nn;
        } else if constexpr (EPI == EPI_TANH_B) {
          outb[(size_t)row * N + col] = (__bf16)tanhf(v);
        } else if constexpr (EPI == EPI_NONE) {
          outf[(size_t)row * N + col] = v;
        } else if constexpr (EPI == EPI_TANH_FB) {
          float m  = tanhf(v + bias[col]);
          size_t o = (size_t)row * N + col;
          outf[o] = m;
          outb[o] = (__bf16)m;
        } else {  // EPI_HBLEND: h update + d_t for next step + sequence out
          float hat  = tanhf(v + bias[col]);
          size_t o   = (size_t)row * Hd + col;
          float z    = zread[o];
          float hold = state[o];
          float hn   = (1.0f - z) * hold + z * hat;
          state[o]  = hn;
          stateb[o] = (__bf16)hn;
          dtb[o]    = (__bf16)(hn - hold);
          seq[(size_t)row * seq_stride + col] = hn;
        }
      }
    }
  }
}

// ---------------- small helper kernels ----------------

__global__ void k_convert_bf16(const float* __restrict__ in,
                               __bf16* __restrict__ out, int n) {
  int i = blockIdx.x * 256 + threadIdx.x;
  if (i < n) out[i] = (__bf16)in[i];
}

// in: [K x N] fp32 row-major  ->  out: [N x K] bf16 row-major (transpose)
__global__ void k_convert_T_bf16(const float* __restrict__ in,
                                 __bf16* __restrict__ out, int K, int N) {
  int i = blockIdx.x * 256 + threadIdx.x;
  if (i < K * N) {
    int k = i / N, n = i % N;
    out[(size_t)n * K + k] = (__bf16)in[i];
  }
}

__global__ void k_zero_f32(float* p, int n) {
  int i = blockIdx.x * 256 + threadIdx.x;
  if (i < n) p[i] = 0.0f;
}

__global__ void k_zero_bf16(__bf16* p, int n) {
  int i = blockIdx.x * 256 + threadIdx.x;
  if (i < n) p[i] = (__bf16)0.0f;
}

// alpha = softmax(score, axis=1); Omega = bf16(alpha * N)   (one block per row)
__global__ __launch_bounds__(256)
void k_softmax_omega(const float* __restrict__ score,
                     const float* __restrict__ Nst,
                     __bf16* __restrict__ omega) {
  __shared__ float red[256];
  int b = blockIdx.x, tid = threadIdx.x;
  float v0 = score[b * Hd + tid];
  float v1 = score[b * Hd + 256 + tid];
  red[tid] = fmaxf(v0, v1);
  __syncthreads();
  for (int s = 128; s > 0; s >>= 1) {
    if (tid < s) red[tid] = fmaxf(red[tid], red[tid + s]);
    __syncthreads();
  }
  float mx = red[0];
  __syncthreads();
  float e0 = __expf(v0 - mx), e1 = __expf(v1 - mx);
  red[tid] = e0 + e1;
  __syncthreads();
  for (int s = 128; s > 0; s >>= 1) {
    if (tid < s) red[tid] += red[tid + s];
    __syncthreads();
  }
  float inv = 1.0f / red[0];
  omega[b * Hd + tid]       = (__bf16)(e0 * inv * Nst[b * Hd + tid]);
  omega[b * Hd + 256 + tid] = (__bf16)(e1 * inv * Nst[b * Hd + 256 + tid]);
}

__global__ void k_final(const float* __restrict__ h,
                        const float* __restrict__ Nst,
                        float* __restrict__ dst) {
  int i = blockIdx.x * 256 + threadIdx.x;       // exactly 262144 threads
  if (i < Bsz * Hd) dst[i] = h[i];
  else              dst[i] = Nst[i - Bsz * Hd];
}

// ---------------- GEMM launch dispatcher ----------------

static void launch_gemm(int epi, hipStream_t stream,
                        const __bf16* A1, int lda1, int K1, const __bf16* WT1,
                        const __bf16* A2, int lda2, int K2, const __bf16* WT2,
                        int Nn, const float* bias,
                        const float* zread, float* zwrite,
                        const float* rmul, __bf16* rbout,
                        float* state, __bf16* stateb, __bf16* dtb,
                        float* outf, __bf16* outb, float* seq, int sstr) {
  dim3 g(Nn / 128, Bsz / 128);
#define GEMM_CALL(E)                                                          \
  wmma_gemm_fused<E><<<g, 256, 0, stream>>>(A1, lda1, K1, WT1, A2, lda2, K2,  \
      WT2, Nn, bias, zread, zwrite, rmul, rbout, state, stateb, dtb, outf,    \
      outb, seq, sstr)
  switch (epi) {
    case EPI_GATE:    GEMM_CALL(EPI_GATE);    break;
    case EPI_NBLEND:  GEMM_CALL(EPI_NBLEND);  break;
    case EPI_TANH_B:  GEMM_CALL(EPI_TANH_B);  break;
    case EPI_NONE:    GEMM_CALL(EPI_NONE);    break;
    case EPI_TANH_FB: GEMM_CALL(EPI_TANH_FB); break;
    default:          GEMM_CALL(EPI_HBLEND);  break;
  }
#undef GEMM_CALL
}

// ---------------- entry point ----------------

extern "C" void kernel_launch(void* const* d_in, const int* in_sizes, int n_in,
                              void* d_out, int out_size, void* d_ws,
                              size_t ws_size, hipStream_t stream) {
  (void)in_sizes; (void)n_in; (void)out_size; (void)ws_size;

  const float* x     = (const float*)d_in[0];
  const float* Wa    = (const float*)d_in[1];
  const float* Ua    = (const float*)d_in[2];
  const float* ba    = (const float*)d_in[3];
  const float* Wna   = (const float*)d_in[4];
  const float* Una   = (const float*)d_in[5];
  const float* bna   = (const float*)d_in[6];
  const float* WAttn = (const float*)d_in[7];
  const float* UAttn = (const float*)d_in[8];
  const float* VAttn = (const float*)d_in[9];
  const float* Wma   = (const float*)d_in[10];
  const float* Uma   = (const float*)d_in[11];
  const float* bma   = (const float*)d_in[12];
  const float* Wm    = (const float*)d_in[13];
  const float* Um    = (const float*)d_in[14];
  const float* bmg   = (const float*)d_in[15];
  const float* Whm   = (const float*)d_in[16];
  const float* Uhm   = (const float*)d_in[17];
  const float* bhm   = (const float*)d_in[18];

  // ---- workspace carve-up (256B aligned) ----
  char*  ws  = (char*)d_ws;
  size_t off = 0;
  auto carve = [&](size_t bytes) -> char* {
    char* p = ws + off;
    off = (off + bytes + 255) & ~(size_t)255;
    return p;
  };
  const int SH = Bsz * Hd;                 // 131072 state elems

  __bf16* xb    = (__bf16*)carve((size_t)Bsz * Tlen * Ein * 2);
  // transposed bf16 weights, [N x K] row-major
  __bf16* WaT   = (__bf16*)carve((size_t)2 * Hd * Hd * 2);
  __bf16* UaT   = (__bf16*)carve((size_t)2 * Hd * Hd * 2);
  __bf16* WnaT  = (__bf16*)carve((size_t)Hd * Hd * 2);
  __bf16* UnaT  = (__bf16*)carve((size_t)Hd * Hd * 2);
  __bf16* WAtT  = (__bf16*)carve((size_t)Hd * Hd * 2);
  __bf16* UAtT  = (__bf16*)carve((size_t)Hd * Hd * 2);
  __bf16* VAtT  = (__bf16*)carve((size_t)Hd * Hd * 2);
  __bf16* WmaT  = (__bf16*)carve((size_t)Hd * Hd * 2);
  __bf16* UmaT  = (__bf16*)carve((size_t)Hd * Hd * 2);
  __bf16* WmT   = (__bf16*)carve((size_t)2 * Hd * Ein * 2);
  __bf16* UmT   = (__bf16*)carve((size_t)2 * Hd * Hd * 2);
  __bf16* WhmT  = (__bf16*)carve((size_t)Hd * Ein * 2);
  __bf16* UhmT  = (__bf16*)carve((size_t)Hd * Hd * 2);

  float*  hF     = (float*)carve((size_t)SH * 4);
  float*  NF     = (float*)carve((size_t)SH * 4);
  float*  zA     = (float*)carve((size_t)SH * 4);   // z_A gate
  float*  zM     = (float*)carve((size_t)SH * 4);   // z_M gate
  float*  scoreF = (float*)carve((size_t)SH * 4);
  float*  mF     = (float*)carve((size_t)SH * 4);

  __bf16* dtB    = (__bf16*)carve((size_t)SH * 2);
  __bf16* hB     = (__bf16*)carve((size_t)SH * 2);
  __bf16* NB     = (__bf16*)carve((size_t)SH * 2);
  __bf16* NrA    = (__bf16*)carve((size_t)SH * 2);
  __bf16* sB     = (__bf16*)carve((size_t)SH * 2);
  __bf16* OmB    = (__bf16*)carve((size_t)SH * 2);
  __bf16* mB     = (__bf16*)carve((size_t)SH * 2);
  __bf16* mrM    = (__bf16*)carve((size_t)SH * 2);

  // ---- one-time conversions: x (bf16), weights (bf16, transposed) ----
  {
    int n = Bsz * Tlen * Ein;
    k_convert_bf16<<<(n + 255) / 256, 256, 0, stream>>>(x, xb, n);
  }
  struct CvtT { const float* s; __bf16* d; int K, N; };
  const CvtT cvt[] = {
    { Wa,    WaT,  Hd,  2 * Hd }, { Ua,   UaT,  Hd,  2 * Hd },
    { Wna,   WnaT, Hd,  Hd },     { Una,  UnaT, Hd,  Hd },
    { WAttn, WAtT, Hd,  Hd },     { UAttn,UAtT, Hd,  Hd },
    { VAttn, VAtT, Hd,  Hd },     { Wma,  WmaT, Hd,  Hd },
    { Uma,   UmaT, Hd,  Hd },     { Wm,   WmT,  Ein, 2 * Hd },
    { Um,    UmT,  Hd,  2 * Hd }, { Whm,  WhmT, Ein, Hd },
    { Uhm,   UhmT, Hd,  Hd },
  };
  for (const CvtT& c : cvt) {
    int n = c.K * c.N;
    k_convert_T_bf16<<<(n + 255) / 256, 256, 0, stream>>>(c.s, c.d, c.K, c.N);
  }

  // ---- zero initial state: h = N = 0, d_t = 0 ----
  k_zero_f32 <<<SH / 256, 256, 0, stream>>>(hF, SH);
  k_zero_f32 <<<SH / 256, 256, 0, stream>>>(NF, SH);
  k_zero_bf16<<<SH / 256, 256, 0, stream>>>(dtB, SH);
  k_zero_bf16<<<SH / 256, 256, 0, stream>>>(hB, SH);
  k_zero_bf16<<<SH / 256, 256, 0, stream>>>(NB, SH);

  float* seq_base = (float*)d_out;
  const int xlda = Tlen * Ein;   // row stride of x_t slices

  for (int t = 0; t < Tlen; ++t) {
    const __bf16* xt = xb + (size_t)t * Ein;
    // 1) gates_A = sigmoid(d_t@Wa + N@Ua + ba): z_A fp32 + NrA = bf16(N*r_A)
    launch_gemm(EPI_GATE, stream, dtB, Hd, Hd, WaT, NB, Hd, Hd, UaT,
                2 * Hd, ba, nullptr, zA, NF, NrA,
                nullptr, nullptr, nullptr, nullptr, nullptr, nullptr, 0);
    // 2) N_new = (1-z_A)N + z_A*tanh(d_t@Wna + NrA@Una + bna)
    launch_gemm(EPI_NBLEND, stream, dtB, Hd, Hd, WnaT, NrA, Hd, Hd, UnaT,
                Hd, bna, zA, nullptr, nullptr, nullptr,
                NF, NB, nullptr, nullptr, nullptr, nullptr, 0);
    // 3) s = tanh(h@WAttn + N_new@UAttn)
    launch_gemm(EPI_TANH_B, stream, hB, Hd, Hd, WAtT, NB, Hd, Hd, UAtT,
                Hd, nullptr, nullptr, nullptr, nullptr, nullptr,
                nullptr, nullptr, nullptr, nullptr, sB, nullptr, 0);
    // 4) score = s @ VAttn
    launch_gemm(EPI_NONE, stream, sB, Hd, Hd, VAtT, nullptr, 0, 0, nullptr,
                Hd, nullptr, nullptr, nullptr, nullptr, nullptr,
                nullptr, nullptr, nullptr, scoreF, nullptr, nullptr, 0);
    //    Omega = softmax(score, axis=1) * N_new
    k_softmax_omega<<<Bsz, 256, 0, stream>>>(scoreF, NF, OmB);
    // 5) m = tanh(Omega@Wma + h@Uma + bma)
    launch_gemm(EPI_TANH_FB, stream, OmB, Hd, Hd, WmaT, hB, Hd, Hd, UmaT,
                Hd, bma, nullptr, nullptr, nullptr, nullptr,
                nullptr, nullptr, nullptr, mF, mB, nullptr, 0);
    // 6) gates_M = sigmoid(x_t@Wm + m@Um + bm): z_M fp32 + mrM = bf16(m*r_M)
    launch_gemm(EPI_GATE, stream, xt, xlda, Ein, WmT, mB, Hd, Hd, UmT,
                2 * Hd, bmg, nullptr, zM, mF, mrM,
                nullptr, nullptr, nullptr, nullptr, nullptr, nullptr, 0);
    // 7) h_new = (1-z_M)h + z_M*tanh(x_t@Whm + mrM@Uhm + bhm); seq + d_t out
    launch_gemm(EPI_HBLEND, stream, xt, xlda, Ein, WhmT, mrM, Hd, Hd, UhmT,
                Hd, bhm, zM, nullptr, nullptr, nullptr,
                hF, hB, dtB, nullptr, nullptr,
                seq_base + (size_t)t * Hd, Tlen * Hd);
  }

  // tail outputs: (h_T, N_T)
  k_final<<<(2 * SH) / 256, 256, 0, stream>>>(
      hF, NF, (float*)d_out + (size_t)Bsz * Tlen * Hd);
}
